// E8QLinear_34651796144741
// MI455X (gfx1250) — matrix operand order repaired and checked by reading, not compiled, not run on hardware
//
#include <hip/hip_runtime.h>
#include <stdint.h>

// ---------------------------------------------------------------------------
// E8-lattice quantized linear layer for MI455X (gfx1250, wave32, WMMA).
//   out = x @ wq.T + bias,  wq = E8quant(w / ||w||_row * sf) * ||w||_row / sf
// Strategy:
//   1) quantize kernel: q (half-integer lattice points, EXACT in f16) + per-row
//      scale (norm/sf) pulled out of the GEMM.
//   2) WMMA GEMM: x split into f16 hi/lo (x = hi + lo, ~22 mantissa bits), both
//      multiplied against exact-f16 q with v_wmma_f32_16x16x32_f16, f32 acc.
//      B tile staged with gfx1250 async global->LDS DMA (ASYNCcnt) overlapping
//      the A-tile fp32->f16 split. Epilogue applies per-column scale and bias.
// ---------------------------------------------------------------------------

typedef __attribute__((ext_vector_type(16))) _Float16 v16h;
typedef __attribute__((ext_vector_type(8)))  _Float16 v8h;
typedef __attribute__((ext_vector_type(4)))  _Float16 v4h;
typedef __attribute__((ext_vector_type(8)))  float    v8f;

#define CAT16(lo, hi) __builtin_shufflevector((lo), (hi), 0,1,2,3,4,5,6,7,8,9,10,11,12,13,14,15)

#if defined(__gfx1250__) && __has_builtin(__builtin_amdgcn_global_load_async_to_lds_b128)
#define USE_ASYNC_LDS 1
// ROCm 7.2 clang-22 signature (from diagnostics):
//   void __builtin_amdgcn_global_load_async_to_lds_b128(
//       int __vector(4) addrspace(1)* gsrc, int __vector(4) addrspace(3)* ldst,
//       imm int offset, imm int cpol)
typedef int v4i_vs __attribute__((vector_size(16)));
typedef __attribute__((address_space(1))) v4i_vs glb_v4i;
typedef __attribute__((address_space(3))) v4i_vs lds_v4i;
#else
#define USE_ASYNC_LDS 0
#endif

// ---------------------------------------------------------------------------
// Kernel 1: per-row E8 lattice quantization (matches jnp semantics exactly:
// rintf == round-half-to-even, argmax picks first max, d>=0 -> +1).
// ---------------------------------------------------------------------------
__device__ __forceinline__ void nearest_d8(const float x[8], float out[8]) {
    float f[8], d[8];
    float sum = 0.0f;
#pragma unroll
    for (int j = 0; j < 8; ++j) {
        f[j] = rintf(x[j]);          // RNE, same as jnp.round
        d[j] = x[j] - f[j];
        sum += f[j];
    }
    const int  isum = (int)sum;      // exact: sum of integers
    const bool even = ((isum & 1) == 0);
    int   idx  = 0;
    float best = fabsf(d[0]);
#pragma unroll
    for (int j = 1; j < 8; ++j) {
        float a = fabsf(d[j]);
        if (a > best) { best = a; idx = j; }   // strict >: first occurrence
    }
#pragma unroll
    for (int j = 0; j < 8; ++j) out[j] = f[j];
    if (!even) out[idx] += (d[idx] >= 0.0f) ? 1.0f : -1.0f;
}

__global__ __launch_bounds__(256) void e8_quant_kernel(
    const float* __restrict__ W, _Float16* __restrict__ Q,
    float* __restrict__ scaleArr, int K, float sf)
{
    __shared__ float red[256];
    const int r = blockIdx.x;
    const int t = threadIdx.x;
    const float* wr = W + (size_t)r * K;

    // row norm
    float ss = 0.0f;
    for (int i = t; i < K; i += 256) { float v = wr[i]; ss += v * v; }
    red[t] = ss;
    __syncthreads();
    for (int s = 128; s > 0; s >>= 1) {
        if (t < s) red[t] += red[t + s];
        __syncthreads();
    }
    const float nsafe = fmaxf(sqrtf(red[0]), 1e-8f);
    const float s = sf / nsafe;
    if (t == 0) scaleArr[r] = nsafe / sf;

    const int ngroups = K >> 3;
    for (int g = t; g < ngroups; g += 256) {
        float e[8], es[8], y0[8], y1[8];
#pragma unroll
        for (int j = 0; j < 8; ++j) e[j] = wr[g * 8 + j] * s;
        nearest_d8(e, y0);
#pragma unroll
        for (int j = 0; j < 8; ++j) es[j] = e[j] - 0.5f;
        nearest_d8(es, y1);
#pragma unroll
        for (int j = 0; j < 8; ++j) y1[j] += 0.5f;
        float d0 = 0.0f, d1 = 0.0f;
#pragma unroll
        for (int j = 0; j < 8; ++j) {
            float a = e[j] - y0[j], b = e[j] - y1[j];
            d0 += a * a; d1 += b * b;
        }
        const bool pick0 = (d0 <= d1);
        _Float16* qp = Q + (size_t)r * K + g * 8;
#pragma unroll
        for (int j = 0; j < 8; ++j)
            qp[j] = (_Float16)(pick0 ? y0[j] : y1[j]);   // exact in f16
    }
}

// ---------------------------------------------------------------------------
// Kernel 2: WMMA GEMM  out[b,o] = (sum_k x[b,k]*q[o,k]) * scale[o] + bias[o]
// Block: 256 threads (8 waves, 2x4), tile 128(M) x 128(N), BK = 32.
// ---------------------------------------------------------------------------
#define TILE_M 128
#define TILE_N 128
#define TILE_K 32
#define LDH    40   // padded halves per LDS row (80B stride, 16B-aligned chunks)

__global__ __launch_bounds__(256) void e8q_wmma_gemm(
    const float* __restrict__ X, const _Float16* __restrict__ Q,
    const float* __restrict__ scaleArr, const float* __restrict__ bias,
    float* __restrict__ out, int K, int N)
{
    __shared__ __align__(16) _Float16 Ah[TILE_M * LDH];
    __shared__ __align__(16) _Float16 Al[TILE_M * LDH];
    __shared__ __align__(16) _Float16 Bs[TILE_N * LDH];

    const int t    = threadIdx.x;
    const int lane = t & 31;
    const int wave = t >> 5;
    const int hl   = (lane >> 4) & 1;   // lane half (wave32)
    const int l16  = lane & 15;

    const int gM0 = blockIdx.y * TILE_M;
    const int gN0 = blockIdx.x * TILE_N;
    const int waveM = (wave >> 2) * 64;  // 0 or 64
    const int waveN = (wave & 3) * 32;   // 0..96

    const v8f vzero = {0.f, 0.f, 0.f, 0.f, 0.f, 0.f, 0.f, 0.f};
    v8f acc[4][2];
#pragma unroll
    for (int i = 0; i < 4; ++i)
#pragma unroll
        for (int j = 0; j < 2; ++j) acc[i][j] = vzero;

    const int kTiles = K / TILE_K;
    for (int kt = 0; kt < kTiles; ++kt) {
        const int k0 = kt * TILE_K;

        // ---- Stage B first: 128x32 f16, async global->LDS DMA (overlaps A) ----
#pragma unroll
        for (int i = 0; i < 2; ++i) {
            const int c    = t + i * 256;      // 0..511 chunks of 8 halves (16B)
            const int row  = c >> 2;
            const int col8 = (c & 3) << 3;
            const _Float16* gsrc = Q + (size_t)(gN0 + row) * K + k0 + col8;
            _Float16*       ldst = &Bs[row * LDH + col8];
#if USE_ASYNC_LDS
            __builtin_amdgcn_global_load_async_to_lds_b128(
                (glb_v4i*)(uintptr_t)gsrc,
                (lds_v4i*)(uint32_t)(uintptr_t)ldst,
                0, 0);
#else
            *(v8h*)ldst = *(const v8h*)gsrc;
#endif
        }

        // ---- Stage A: 128x32 fp32; issue all loads, then convert hi/lo ----
        float4 av[4];
        int    arowi[4], acoli[4];
#pragma unroll
        for (int i = 0; i < 4; ++i) {
            const int c = t + i * 256;         // 0..1023 float4 chunks
            arowi[i] = c >> 3;
            acoli[i] = (c & 7) << 2;
            av[i] = *(const float4*)(X + (size_t)(gM0 + arowi[i]) * K + k0 + acoli[i]);
        }
#pragma unroll
        for (int i = 0; i < 4; ++i) {
            v4h hi, lo;
            hi.x = (_Float16)av[i].x; lo.x = (_Float16)(av[i].x - (float)hi.x);
            hi.y = (_Float16)av[i].y; lo.y = (_Float16)(av[i].y - (float)hi.y);
            hi.z = (_Float16)av[i].z; lo.z = (_Float16)(av[i].z - (float)hi.z);
            hi.w = (_Float16)av[i].w; lo.w = (_Float16)(av[i].w - (float)hi.w);
            *(v4h*)&Ah[arowi[i] * LDH + acoli[i]] = hi;
            *(v4h*)&Al[arowi[i] * LDH + acoli[i]] = lo;
        }

        // prefetch next K-tile of X while we compute this one
        if (kt + 1 < kTiles) {
            const int prow = t >> 1, pcol = (t & 1) << 4;
            __builtin_prefetch(X + (size_t)(gM0 + prow) * K + k0 + TILE_K + pcol, 0, 1);
        }

#if USE_ASYNC_LDS
        // each wave waits for its own async DMA before the workgroup barrier
#if __has_builtin(__builtin_amdgcn_s_wait_asynccnt)
        __builtin_amdgcn_s_wait_asynccnt(0);
#else
        asm volatile("s_wait_asynccnt 0" ::: "memory");
#endif
#endif
        __syncthreads();

        // ---- B fragments for this wave (lane<16: K 0-7 & 16-23; lane>=16: 8-15 & 24-31) ----
        v16h bfrag[2];
#pragma unroll
        for (int nf = 0; nf < 2; ++nf) {
            const _Float16* bp = &Bs[(waveN + nf * 16 + l16) * LDH + hl * 8];
            v8h b0 = *(const v8h*)bp;
            v8h b1 = *(const v8h*)(bp + 16);
            bfrag[nf] = CAT16(b0, b1);
        }
        // ---- A fragments + WMMA (lane<16: K 0-7 & 16-23; lane>=16: 8-15 & 24-31) ----
#pragma unroll
        for (int mf = 0; mf < 4; ++mf) {
            const int arow = (waveM + mf * 16 + l16) * LDH;
            const int c0   = hl * 8;
            v8h a0 = *(const v8h*)&Ah[arow + c0];
            v8h a1 = *(const v8h*)&Ah[arow + c0 + 16];
            v16h ahi = CAT16(a0, a1);
            v8h a2 = *(const v8h*)&Al[arow + c0];
            v8h a3 = *(const v8h*)&Al[arow + c0 + 16];
            v16h alo = CAT16(a2, a3);
#pragma unroll
            for (int nf = 0; nf < 2; ++nf) {
                acc[mf][nf] = __builtin_amdgcn_wmma_f32_16x16x32_f16(
                    false, ahi, false, bfrag[nf], (short)0, acc[mf][nf], false, false);
                acc[mf][nf] = __builtin_amdgcn_wmma_f32_16x16x32_f16(
                    false, alo, false, bfrag[nf], (short)0, acc[mf][nf], false, false);
            }
        }
        __syncthreads();
    }

    // ---- Epilogue: C layout => lane<16: M=v, lane>=16: M=v+8; N = lane&15 ----
#pragma unroll
    for (int nf = 0; nf < 2; ++nf) {
        const int col = gN0 + waveN + nf * 16 + l16;
        const float sc = scaleArr[col];
        const float bi = bias[col];
#pragma unroll
        for (int mf = 0; mf < 4; ++mf) {
            const int rbase = gM0 + waveM + mf * 16 + hl * 8;
#pragma unroll
            for (int v = 0; v < 8; ++v) {
                out[(size_t)(rbase + v) * N + col] = acc[mf][nf][v] * sc + bi;
            }
        }
    }
}

// ---------------------------------------------------------------------------
// Host launcher
// ---------------------------------------------------------------------------
extern "C" void kernel_launch(void* const* d_in, const int* in_sizes, int n_in,
                              void* d_out, int out_size, void* d_ws, size_t ws_size,
                              hipStream_t stream)
{
    const float* x    = (const float*)d_in[0];
    const float* w    = (const float*)d_in[1];
    const float* bias = (const float*)d_in[2];
    float*       out  = (float*)d_out;

    const int OUT  = in_sizes[2];
    const int K    = in_sizes[1] / OUT;
    const int Bdim = in_sizes[0] / K;

    // adaptive = min(2, max(0.5, n_rows/10000)); sf = 60 * adaptive
    float adaptive = (float)OUT / 10000.0f;
    if (adaptive < 0.5f) adaptive = 0.5f;
    if (adaptive > 2.0f) adaptive = 2.0f;
    const float sf = 60.0f * adaptive;

    // workspace layout: [scale: OUT floats, 256B aligned pad][Q: OUT*K f16]
    float* scaleArr = (float*)d_ws;
    size_t scaleBytes = ((size_t)OUT * sizeof(float) + 255) & ~(size_t)255;
    _Float16* Q = (_Float16*)((char*)d_ws + scaleBytes);

    e8_quant_kernel<<<OUT, 256, 0, stream>>>(w, Q, scaleArr, K, sf);

    dim3 grid(OUT / TILE_N, Bdim / TILE_M);
    e8q_wmma_gemm<<<grid, 256, 0, stream>>>(x, Q, scaleArr, bias, out, K, OUT);
}
